// ScaledDotProductAttention_68015102099779
// MI455X (gfx1250) — matrix operand order repaired and checked by reading, not compiled, not run on hardware
//
#include <hip/hip_runtime.h>

// Flash-attention (with full attn-matrix output) for B=32, L=2048, D=64, fp32 IO.
// WMMA f16 path: v_wmma_f32_16x16x32_f16, wave32, one 16-row Q tile per wave.
// d_ws holds preconverted f16 K (row-major) and f16 V^T ([b][d][key]); hot loops
// issue pure b128 f16 loads. Q is pre-scaled by 1/sqrt(64) so no per-score mul.
// Independent S-tile pairs per iteration hide WMMA->WMMA/VALU hazard slots.

typedef __attribute__((ext_vector_type(16))) _Float16 v16h;
typedef __attribute__((ext_vector_type(8)))  _Float16 v8h;
typedef __attribute__((ext_vector_type(4)))  _Float16 v4h;
typedef __attribute__((ext_vector_type(8)))  float    v8f;
typedef __attribute__((ext_vector_type(4)))  float    v4f;

#define BATCH 32
#define LSEQ  2048
#define DDIM  64
#define QTILE 16
#define WAVES_PER_BLOCK 4

#define WMMA_F16(A, B, C) \
  __builtin_amdgcn_wmma_f32_16x16x32_f16(false, (A), false, (B), (short)0, (C), false, false)

__device__ __forceinline__ v16h cvt16(const float* f) {
  v16h h;
#pragma unroll
  for (int j = 0; j < 16; ++j) h[j] = (_Float16)f[j];
  return h;
}

// 16 contiguous f16 -> v16h (two b128 loads)
__device__ __forceinline__ v16h load_h16(const _Float16* p) {
  v8h a = *(const v8h*)p;
  v8h b = *(const v8h*)(p + 8);
  v16h r;
#pragma unroll
  for (int j = 0; j < 8; ++j) { r[j] = a[j]; r[j + 8] = b[j]; }
  return r;
}

// A-fragment (16x32 f16) from a row-major f32 row, scaled: halves 0-15 =
// [cb+2*klo..cb+2*klo+15]  (klo = 8*(lane>=16))
__device__ __forceinline__ v16h load_a_row_scaled(const float* rowptr, int cb,
                                                  int klo, float scale) {
  float t[16];
  *(v4f*)(t + 0)  = *(const v4f*)(rowptr + cb + 2 * klo);
  *(v4f*)(t + 4)  = *(const v4f*)(rowptr + cb + 2 * klo + 4);
  *(v4f*)(t + 8)  = *(const v4f*)(rowptr + cb + 2 * klo + 8);
  *(v4f*)(t + 12) = *(const v4f*)(rowptr + cb + 2 * klo + 12);
  v16h h;
#pragma unroll
  for (int j = 0; j < 16; ++j) h[j] = (_Float16)(t[j] * scale);
  return h;
}

// B-fragment (32x16 f16): 16 contiguous f32 from one K row, converted
__device__ __forceinline__ v16h load_b_row_f32(const float* p) {
  float t[16];
#pragma unroll
  for (int j = 0; j < 4; ++j) *(v4f*)(t + 4 * j) = *(const v4f*)(p + 4 * j);
  return cvt16(t);
}

// ---------------- preconversion kernels (use d_ws) ---------------------------
__global__ __launch_bounds__(256) void convert_k_f16(const float* __restrict__ k,
                                                     _Float16* __restrict__ k16) {
  size_t i = ((size_t)blockIdx.x * 256 + threadIdx.x) * 4;  // BATCH*LSEQ*DDIM / 4 threads
  v4f x = *(const v4f*)(k + i);
  v4h h;
#pragma unroll
  for (int j = 0; j < 4; ++j) h[j] = (_Float16)x[j];
  *(v4h*)(k16 + i) = h;
}

__global__ __launch_bounds__(256) void convert_v_t16(const float* __restrict__ v,
                                                     _Float16* __restrict__ vt) {
  // vt[b][d][key0..key0+3] = v[b][key][d]
  size_t tid = (size_t)blockIdx.x * 256 + threadIdx.x;      // BATCH*DDIM*(LSEQ/4) threads
  int b  = (int)(tid / (DDIM * (LSEQ / 4)));
  int r  = (int)(tid % (DDIM * (LSEQ / 4)));
  int d  = r / (LSEQ / 4);
  int k0 = (r % (LSEQ / 4)) * 4;
  v4h h;
#pragma unroll
  for (int j = 0; j < 4; ++j)
    h[j] = (_Float16)v[((size_t)b * LSEQ + k0 + j) * DDIM + d];
  *(v4h*)(vt + ((size_t)b * DDIM + d) * LSEQ + k0) = h;
}

// ---------------- main attention kernel --------------------------------------
template <bool USE_F16>
__global__ __launch_bounds__(WAVES_PER_BLOCK * 32)
void attn_main(const float* __restrict__ q, const float* __restrict__ k,
               const float* __restrict__ v, const _Float16* __restrict__ k16,
               const _Float16* __restrict__ vt16, float* __restrict__ out,
               float* __restrict__ attn) {
  __shared__ _Float16 plds_all[WAVES_PER_BLOCK][QTILE * 32];

  const int lane  = threadIdx.x & 31;
  const int wave  = threadIdx.x >> 5;
  const int gtile = blockIdx.x * WAVES_PER_BLOCK + wave;  // 4096 q-tiles total
  const int b     = gtile >> 7;                           // 128 tiles per batch
  const int q0    = (gtile & 127) * QTILE;

  const int n     = lane & 15;   // B/C/D column index; also A row index
  const int hi    = lane >> 4;
  const int klo   = hi * 8;      // A-layout K offset
  const int kbase = hi * 16;     // B-layout K offset

  const float*    qb    = q + (size_t)b * LSEQ * DDIM;
  const float*    kbp   = k + (size_t)b * LSEQ * DDIM;
  const float*    vbp   = v + (size_t)b * LSEQ * DDIM;
  const _Float16* k16b  = USE_F16 ? (k16 + (size_t)b * LSEQ * DDIM) : nullptr;
  const _Float16* vt16b = USE_F16 ? (vt16 + (size_t)b * DDIM * LSEQ) : nullptr;

  // Q fragments pre-scaled by 1/sqrt(64): S = (Q/8) . K^T needs no later scale
  const float* qrow = qb + (size_t)(q0 + n) * DDIM;
  const v16h a_q0 = load_a_row_scaled(qrow, 0, klo, 0.125f);
  const v16h a_q1 = load_a_row_scaled(qrow, 32, klo, 0.125f);

  // ---------------- Pass 1: online row max / sum (32 keys/iter, 2 indep S) ---
  float mrow[8], lrow[8];
#pragma unroll
  for (int r = 0; r < 8; ++r) { mrow[r] = -3.0e38f; lrow[r] = 0.0f; }

#pragma unroll 1
  for (int n0 = 0; n0 < LSEQ; n0 += 32) {
    v16h b00, b01, b10, b11;
    if constexpr (USE_F16) {
      const _Float16* kr0 = k16b + (size_t)(n0 + n) * DDIM + kbase;
      const _Float16* kr1 = kr0 + 16 * DDIM;
      __builtin_prefetch(kr0 + 32 * DDIM, 0, 1);
      b00 = load_h16(kr0); b01 = load_h16(kr0 + 32);
      b10 = load_h16(kr1); b11 = load_h16(kr1 + 32);
    } else {
      const float* kr0 = kbp + (size_t)(n0 + n) * DDIM + kbase;
      const float* kr1 = kr0 + 16 * DDIM;
      b00 = load_b_row_f32(kr0); b01 = load_b_row_f32(kr0 + 32);
      b10 = load_b_row_f32(kr1); b11 = load_b_row_f32(kr1 + 32);
    }
    v8f s0 = {}, s1 = {};
    s0 = WMMA_F16(a_q0, b00, s0);           // independent pair hides
    s1 = WMMA_F16(a_q0, b10, s1);           // WMMA->WMMA hazard slots
    s0 = WMMA_F16(a_q1, b01, s0);
    s1 = WMMA_F16(a_q1, b11, s1);
#pragma unroll
    for (int r = 0; r < 8; ++r) {
      float mn = fmaxf(mrow[r], fmaxf(s0[r], s1[r]));
      lrow[r] = lrow[r] * __expf(mrow[r] - mn) + __expf(s0[r] - mn) +
                __expf(s1[r] - mn);
      mrow[r] = mn;
    }
  }
  // combine per-column partials across the 16 lanes of each half-wave
#pragma unroll
  for (int r = 0; r < 8; ++r) {
#pragma unroll
    for (int off = 1; off <= 8; off <<= 1) {
      float mo = __shfl_xor(mrow[r], off, 32);
      float lo = __shfl_xor(lrow[r], off, 32);
      float mn = fmaxf(mrow[r], mo);
      lrow[r] = lrow[r] * __expf(mrow[r] - mn) + lo * __expf(mo - mn);
      mrow[r] = mn;
    }
  }
  // p = exp(s - m)/l == exp(s - c), c = m + ln(l)  (no per-element multiply)
  float crow[8];
#pragma unroll
  for (int r = 0; r < 8; ++r) crow[r] = mrow[r] + __logf(lrow[r]);

  // ---------------- Pass 2: recompute S, emit attn (NT), accumulate O --------
  _Float16* plds = plds_all[wave];
  v8f acc[4] = {};  // O tiles: D columns [16t, 16t+16)
  float* attn_base = attn + ((size_t)b * LSEQ + q0) * LSEQ;

#pragma unroll 1
  for (int kb0 = 0; kb0 < LSEQ; kb0 += 32) {
    v16h b00, b01, b10, b11;
    if constexpr (USE_F16) {
      const _Float16* kr0 = k16b + (size_t)(kb0 + n) * DDIM + kbase;
      const _Float16* kr1 = kr0 + 16 * DDIM;
      __builtin_prefetch(kr0 + 32 * DDIM, 0, 1);
      b00 = load_h16(kr0); b01 = load_h16(kr0 + 32);
      b10 = load_h16(kr1); b11 = load_h16(kr1 + 32);
    } else {
      const float* kr0 = kbp + (size_t)(kb0 + n) * DDIM + kbase;
      const float* kr1 = kr0 + 16 * DDIM;
      b00 = load_b_row_f32(kr0); b01 = load_b_row_f32(kr0 + 32);
      b10 = load_b_row_f32(kr1); b11 = load_b_row_f32(kr1 + 32);
    }
    v8f s0 = {}, s1 = {};
    s0 = WMMA_F16(a_q0, b00, s0);
    s1 = WMMA_F16(a_q0, b10, s1);
    s0 = WMMA_F16(a_q1, b01, s0);
    s1 = WMMA_F16(a_q1, b11, s1);
    // normalized probabilities in C/D layout -> LDS tile [row][key] (f16)
#pragma unroll
    for (int r = 0; r < 8; ++r) {
      plds[(r + 8 * hi) * 32 + n]      = (_Float16)__expf(s0[r] - crow[r]);
      plds[(r + 8 * hi) * 32 + 16 + n] = (_Float16)__expf(s1[r] - crow[r]);
    }
    // transpose to A-layout: lane row m=n, keys 2*klo+0..15
    v8h plo = *(const v8h*)(plds + n * 32 + 2 * klo);
    v8h phi = *(const v8h*)(plds + n * 32 + 2 * klo + 8);
    v16h a_p;
#pragma unroll
    for (int j = 0; j < 8; ++j) { a_p[j] = plo[j]; a_p[j + 8] = phi[j]; }

    // stream attn rows (non-temporal: 537 MB write-once, keep L2 for K/V)
    float* ar = attn_base + (size_t)n * LSEQ + kb0;
    v4f st0, st1, st2, st3;
#pragma unroll
    for (int j = 0; j < 4; ++j) {
      st0[j] = (float)a_p[j];
      st1[j] = (float)a_p[j + 4];
      st2[j] = (float)a_p[j + 8];
      st3[j] = (float)a_p[j + 12];
    }
    __builtin_nontemporal_store(st0, (v4f*)(ar + 2 * klo));
    __builtin_nontemporal_store(st1, (v4f*)(ar + 2 * klo + 4));
    __builtin_nontemporal_store(st2, (v4f*)(ar + 2 * klo + 8));
    __builtin_nontemporal_store(st3, (v4f*)(ar + 2 * klo + 12));

    // O += P(16x32) @ V(32x64): 4 column tiles
#pragma unroll
    for (int t = 0; t < 4; ++t) {
      v16h bv;
      if constexpr (USE_F16) {
        // V^T layout: row = d-column, 16 contiguous keys -> 2 x b128
        const _Float16* vr = vt16b + (size_t)(t * 16 + n) * LSEQ + kb0 + kbase;
        __builtin_prefetch(vr + 32, 0, 1);
        bv = load_h16(vr);
      } else {
        float tv[16];
#pragma unroll
        for (int j = 0; j < 16; ++j)
          tv[j] = vbp[(size_t)(kb0 + kbase + j) * DDIM + t * 16 + n];
        bv = cvt16(tv);
      }
      acc[t] = WMMA_F16(a_p, bv, acc[t]);
    }
  }

  // ---------------- store O (f32, C/D layout) --------------------------------
  float* ob = out + ((size_t)b * LSEQ + q0) * DDIM;
#pragma unroll
  for (int t = 0; t < 4; ++t)
#pragma unroll
    for (int r = 0; r < 8; ++r)
      ob[(size_t)(r + 8 * hi) * DDIM + t * 16 + n] = acc[t][r];
}

extern "C" void kernel_launch(void* const* d_in, const int* in_sizes, int n_in,
                              void* d_out, int out_size, void* d_ws, size_t ws_size,
                              hipStream_t stream) {
  (void)in_sizes; (void)n_in; (void)out_size;
  const float* q = (const float*)d_in[0];
  const float* k = (const float*)d_in[1];
  const float* v = (const float*)d_in[2];
  float* out  = (float*)d_out;                                  // [B, L, D]
  float* attn = out + (size_t)BATCH * LSEQ * DDIM;              // [B, L, L]

  const size_t nkv = (size_t)BATCH * LSEQ * DDIM;               // 4,194,304
  const size_t ws_needed = 2 * nkv * sizeof(_Float16);          // 16 MB

  const int qtiles = BATCH * (LSEQ / QTILE);                    // 4096
  dim3 grid(qtiles / WAVES_PER_BLOCK);                          // 1024
  dim3 block(WAVES_PER_BLOCK * 32);                             // 128 (4 waves)

  if (ws_size >= ws_needed) {
    _Float16* k16 = (_Float16*)d_ws;
    _Float16* vt  = k16 + nkv;
    convert_k_f16<<<dim3((unsigned)(nkv / 4 / 256)), dim3(256), 0, stream>>>(k, k16);
    convert_v_t16<<<dim3((unsigned)(nkv / 4 / 256)), dim3(256), 0, stream>>>(v, vt);
    attn_main<true><<<grid, block, 0, stream>>>(q, k, v, k16, vt, out, attn);
  } else {
    attn_main<false><<<grid, block, 0, stream>>>(q, k, v, nullptr, nullptr, out, attn);
  }
}
